// MultiHeadSelfAttention_13228499271978
// MI455X (gfx1250) — matrix operand order repaired and checked by reading, not compiled, not run on hardware
//
#include <hip/hip_runtime.h>
#include <stddef.h>

// ---------------------------------------------------------------------------
// Problem constants (from reference)
// ---------------------------------------------------------------------------
#define D_MODEL   1024
#define NUM_HEADS 16
#define DK        64
#define SEQ       2048
#define BATCH     2
#define M_TOT     (BATCH * SEQ)   // 4096 rows total

typedef __attribute__((ext_vector_type(16))) __bf16 v16bf;
typedef __attribute__((ext_vector_type(8)))  float  v8f;

union Frag {
    v16bf v;
    uint4 q[2];
};

__device__ __forceinline__ unsigned short f2bf(float f) {
    union { float f; unsigned u; } c; c.f = f;
    unsigned u = c.u;
    u += 0x7FFFu + ((u >> 16) & 1u);   // round-to-nearest-even
    return (unsigned short)(u >> 16);
}
__device__ __forceinline__ float bf2f(unsigned short h) {
    union { unsigned u; float f; } c; c.u = ((unsigned)h) << 16; return c.f;
}

__device__ __forceinline__ v8f zero8() {
    v8f z = {0.f, 0.f, 0.f, 0.f, 0.f, 0.f, 0.f, 0.f};
    return z;
}

// ---------------------------------------------------------------------------
// gfx1250 async memory->LDS copy (ASYNCcnt-tracked DMA path, ISA ch.8 §4).
// Builtin prototype (per hipcc diagnostic): the b128 variant takes pointers
// to 128-bit int vectors: (v4i addrspace(1)*, v4i addrspace(3)*, imm, imm).
// Guarded: if this toolchain doesn't declare the builtins, fall back to the
// VGPR round-trip so the file always compiles.
// ---------------------------------------------------------------------------
#if __has_builtin(__builtin_amdgcn_global_load_async_to_lds_b128) && \
    __has_builtin(__builtin_amdgcn_s_wait_asynccnt)
#define HAVE_ASYNC_LDS 1
typedef int v4i __attribute__((vector_size(16)));
typedef __attribute__((address_space(1))) v4i as1_v4i;
typedef __attribute__((address_space(3))) v4i as3_v4i;
#else
#define HAVE_ASYNC_LDS 0
#endif

__device__ __forceinline__ void async_cp_b128(const unsigned short* g,
                                              unsigned short* l) {
#if HAVE_ASYNC_LDS
    __builtin_amdgcn_global_load_async_to_lds_b128((as1_v4i*)g, (as3_v4i*)l,
                                                   0, 0);
#else
    *(uint4*)l = *(const uint4*)g;
#endif
}
__device__ __forceinline__ void async_cp_wait() {
#if HAVE_ASYNC_LDS
    __builtin_amdgcn_s_wait_asynccnt(0);
#endif
}

// ---------------------------------------------------------------------------
// f32 -> bf16 bulk convert (vectorized float4 -> ushort4)
// ---------------------------------------------------------------------------
__global__ void cvt_bf16_kernel(const float* __restrict__ src,
                                unsigned short* __restrict__ dst, int n4) {
    int i = blockIdx.x * blockDim.x + threadIdx.x;
    if (i < n4) {
        float4 f = ((const float4*)src)[i];
        ushort4 o;
        o.x = f2bf(f.x); o.y = f2bf(f.y); o.z = f2bf(f.z); o.w = f2bf(f.w);
        ((ushort4*)dst)[i] = o;
    }
}

// ---------------------------------------------------------------------------
// NT GEMM: C[m,n] = sum_k A[m,k] * W[n,k]     (M=4096, N=1024, K=1024)
// Block tile 128(M) x 64(N), K-step 32.  8 waves: 4(M) x 2(N), 32x32 per wave.
// blockIdx.z selects among up to 3 weight/output pairs (QKV fused launch).
// Tiles staged memory->LDS via gfx1250 async DMA when available.
// ---------------------------------------------------------------------------
template <int OUTF32>
__global__ __launch_bounds__(256) void gemm_nt_bf16(
    const unsigned short* __restrict__ A,
    const unsigned short* __restrict__ W0, const unsigned short* __restrict__ W1,
    const unsigned short* __restrict__ W2,
    unsigned short* __restrict__ Cb0, unsigned short* __restrict__ Cb1,
    unsigned short* __restrict__ Cb2,
    float* __restrict__ Cf) {
    const int KDIM = D_MODEL, NDIM = D_MODEL;
    const unsigned short* W = (blockIdx.z == 0) ? W0 : (blockIdx.z == 1 ? W1 : W2);
    unsigned short* Cb = (blockIdx.z == 0) ? Cb0 : (blockIdx.z == 1 ? Cb1 : Cb2);

    __shared__ __align__(16) unsigned short As[128 * 32];  // 8 KB
    __shared__ __align__(16) unsigned short Bs[64 * 32];   // 4 KB

    const int t = threadIdx.x;
    const int lane = t & 31, wid = t >> 5;
    const int lh = lane >> 4, l15 = lane & 15;     // wave32 half / sub-lane
    const int waveM = wid >> 1, waveN = wid & 1;
    const int wgM = blockIdx.y * 128, wgN = blockIdx.x * 64;

    // per-thread staging coordinates (constant over K loop)
    const int arow = t >> 1, ach = (t & 1) * 16;   // A: 16 bf16 = 2x b128/thread
    const int brow = t >> 2, bch = (t & 3) * 8;    // W:  8 bf16 = 1x b128/thread
    const unsigned short* gA = A + (size_t)(wgM + arow) * KDIM + ach;
    const unsigned short* gW = W + (size_t)(wgN + brow) * KDIM + bch;
    unsigned short* sA = As + arow * 32 + ach;
    unsigned short* sW = Bs + brow * 32 + bch;

    v8f acc[2][2];
    acc[0][0] = zero8(); acc[0][1] = zero8();
    acc[1][0] = zero8(); acc[1][1] = zero8();

    for (int k0 = 0; k0 < KDIM; k0 += 32) {
        // --- async stage: A 128x32, W 64x32 (bf16) ---
        async_cp_b128(gA + k0, sA);
        async_cp_b128(gA + k0 + 8, sA + 8);
        async_cp_b128(gW + k0, sW);
        if (k0 + 32 < KDIM) {  // L2 prefetch of next K tile
            __builtin_prefetch(gA + k0 + 32, 0, 3);
            __builtin_prefetch(gW + k0 + 32, 0, 3);
        }
        async_cp_wait();
        __syncthreads();

        // --- fragments per ISA VGPR layouts ---
        Frag aF[2], bF[2];
#pragma unroll
        for (int mi = 0; mi < 2; ++mi) {
            const uint4* p = (const uint4*)(As + (waveM * 32 + mi * 16 + l15) * 32);
            aF[mi].q[0] = p[lh];          // K = lh*8 .. lh*8+7
            aF[mi].q[1] = p[2 + lh];      // K = 16+lh*8 ..
        }
#pragma unroll
        for (int ni = 0; ni < 2; ++ni) {
            const uint4* p = (const uint4*)(Bs + (waveN * 32 + ni * 16 + l15) * 32);
            bF[ni].q[0] = p[lh];          // K = lh*8 .. lh*8+7
            bF[ni].q[1] = p[2 + lh];      // K = 16+lh*8 ..
        }
#pragma unroll
        for (int mi = 0; mi < 2; ++mi)
#pragma unroll
            for (int ni = 0; ni < 2; ++ni)
                acc[mi][ni] = __builtin_amdgcn_wmma_f32_16x16x32_bf16(
                    false, aF[mi].v, false, bF[ni].v, (short)0, acc[mi][ni],
                    false, false);
        __syncthreads();
    }

    // --- epilogue: C layout = lane col, VGPR r -> row r + 8*half ---
#pragma unroll
    for (int mi = 0; mi < 2; ++mi) {
#pragma unroll
        for (int ni = 0; ni < 2; ++ni) {
            int col = wgN + waveN * 32 + ni * 16 + l15;
#pragma unroll
            for (int r = 0; r < 8; ++r) {
                int row = wgM + waveM * 32 + mi * 16 + 8 * lh + r;
                float v = acc[mi][ni][r];
                if (OUTF32)
                    Cf[(size_t)row * NDIM + col] = v;
                else
                    Cb[(size_t)row * NDIM + col] = f2bf(v);
            }
        }
    }
}

// ---------------------------------------------------------------------------
// RoPE on Q and K (bf16 in, bf16 out). One thread per (row, head, pair).
// ---------------------------------------------------------------------------
__global__ void rope_kernel(const unsigned short* __restrict__ Qp,
                            const unsigned short* __restrict__ Kp,
                            unsigned short* __restrict__ Qr,
                            unsigned short* __restrict__ Kr,
                            const int* __restrict__ pos) {
    int tid = blockIdx.x * blockDim.x + threadIdx.x;
    int i = tid & 31;          // rotary pair index 0..31
    int h = (tid >> 5) & 15;   // head
    int m = tid >> 9;          // flat row (b*S + s)
    if (m >= M_TOT) return;

    float p = (float)pos[m];
    // freq = 10000^{-(2i)/64} = exp(-(2i/64) * ln(10000))
    float freq = __expf(-((float)(2 * i) * (1.0f / 64.0f)) * 9.2103403719761836f);
    float ang = p * freq;
    float s, c;
    __sincosf(ang, &s, &c);

    size_t base = (size_t)m * D_MODEL + h * DK + 2 * i;
    float q1 = bf2f(Qp[base]), q2 = bf2f(Qp[base + 1]);
    Qr[base]     = f2bf(q1 * c - q2 * s);
    Qr[base + 1] = f2bf(q1 * s + q2 * c);
    float k1 = bf2f(Kp[base]), k2 = bf2f(Kp[base + 1]);
    Kr[base]     = f2bf(k1 * c - k2 * s);
    Kr[base + 1] = f2bf(k1 * s + k2 * c);
}

// ---------------------------------------------------------------------------
// Fused causal flash attention, one (128 q-rows, head, batch) per workgroup.
// 8 waves x 16 q-rows; Bc = 64 KV tile shared in LDS; V staged transposed.
// All S x S traffic stays in LDS/registers (never touches HBM).
// K tile streamed memory->LDS with async DMA, overlapped with the VGPR
// transpose-scatter of the V tile.
// ---------------------------------------------------------------------------
__global__ __launch_bounds__(256) void attn_kernel(
    const unsigned short* __restrict__ Q,   // RoPE'd, bf16 [4096][1024]
    const unsigned short* __restrict__ K,   // RoPE'd, bf16
    const unsigned short* __restrict__ V,   // bf16
    unsigned short* __restrict__ O) {       // bf16 out [4096][1024]
    __shared__ __align__(16) unsigned short Ks[64 * 64];       // 8 KB
    __shared__ __align__(16) unsigned short Vst[64 * 64];      // 8 KB (transposed)
    __shared__ __align__(16) unsigned short Ps[8][16 * 64];    // 16 KB (per-wave P)

    const int t = threadIdx.x;
    const int lane = t & 31, wid = t >> 5;
    const int lh = lane >> 4, l15 = lane & 15;
    const int h = blockIdx.y, b = blockIdx.z;
    const int wgQ = blockIdx.x * 128;
    const int qBase = wgQ + wid * 16;
    const size_t batchRow = (size_t)b * SEQ;
    const float scale = 0.125f;  // 1/sqrt(64)

    // Q fragments (loop invariant): A-frag rows = l15, 2 K-steps of 32
    Frag qF[2];
    {
        const uint4* p =
            (const uint4*)(Q + (batchRow + qBase + l15) * D_MODEL + h * DK);
        qF[0].q[0] = p[lh];     qF[0].q[1] = p[2 + lh];
        qF[1].q[0] = p[4 + lh]; qF[1].q[1] = p[6 + lh];
    }

    float mrun[8], lrun[8];
    v8f oacc[4];
#pragma unroll
    for (int r = 0; r < 8; ++r) { mrun[r] = -1e30f; lrun[r] = 0.f; }
#pragma unroll
    for (int dt = 0; dt < 4; ++dt) oacc[dt] = zero8();

    // staging coordinates: 64 rows x 64 cols, 16 elements per thread
    const int srow = t >> 2, scc = (t & 3) * 16;

    const int nTiles = wgQ / 64 + 2;  // causal: KV up to end of this q block
    for (int kt = 0; kt < nTiles; ++kt) {
        const int kvBase = kt * 64;
        // --- K tile: async DMA straight into LDS ---
        {
            const unsigned short* gk =
                K + (batchRow + kvBase + srow) * D_MODEL + h * DK + scc;
            unsigned short* sk = Ks + srow * 64 + scc;
            async_cp_b128(gk, sk);
            async_cp_b128(gk + 8, sk + 8);
        }
        // --- V tile: VGPR load + transpose-scatter (overlaps K DMA) ---
        {
            const uint4* gv = (const uint4*)(V + (batchRow + kvBase + srow) * D_MODEL +
                                             h * DK + scc);
            uint4 v0 = gv[0], v1 = gv[1];
            unsigned short tmp[16];
            *(uint4*)&tmp[0] = v0; *(uint4*)&tmp[8] = v1;
#pragma unroll
            for (int j = 0; j < 16; ++j)
                Vst[(scc + j) * 64 + srow] = tmp[j];  // Vst[d][kv]
        }
        async_cp_wait();
        __syncthreads();

        if (qBase + 15 >= kvBase) {  // wave has unmasked work in this KV tile
            // ---- S = Q K^T (4 col tiles x 2 K-steps = 8 WMMAs) ----
            v8f sacc[4];
#pragma unroll
            for (int tc = 0; tc < 4; ++tc) sacc[tc] = zero8();
#pragma unroll
            for (int ks = 0; ks < 2; ++ks) {
#pragma unroll
                for (int tc = 0; tc < 4; ++tc) {
                    Frag bF;
                    const uint4* p = (const uint4*)(Ks + (tc * 16 + l15) * 64);
                    bF.q[0] = p[ks * 4 + lh];
                    bF.q[1] = p[ks * 4 + 2 + lh];
                    sacc[tc] = __builtin_amdgcn_wmma_f32_16x16x32_bf16(
                        false, qF[ks].v, false, bF.v, (short)0, sacc[tc],
                        false, false);
                }
            }
            // ---- online softmax (rows live in VGPR slots, cols across 16 lanes) ----
#pragma unroll
            for (int r = 0; r < 8; ++r) {
                int qr = qBase + 8 * lh + r;
                float mx = -1e30f;
#pragma unroll
                for (int tc = 0; tc < 4; ++tc) {
                    float s = sacc[tc][r] * scale;
                    int kv = kvBase + tc * 16 + l15;
                    if (kv > qr) s = -1e30f;  // causal mask
                    sacc[tc][r] = s;
                    mx = fmaxf(mx, s);
                }
                mx = fmaxf(mx, __shfl_xor(mx, 1, 16));
                mx = fmaxf(mx, __shfl_xor(mx, 2, 16));
                mx = fmaxf(mx, __shfl_xor(mx, 4, 16));
                mx = fmaxf(mx, __shfl_xor(mx, 8, 16));
                float mnew = fmaxf(mrun[r], mx);
                float corr = __expf(mrun[r] - mnew);
                float rsum = 0.f;
#pragma unroll
                for (int tc = 0; tc < 4; ++tc) {
                    float pv = __expf(sacc[tc][r] - mnew);
                    sacc[tc][r] = pv;
                    rsum += pv;
                }
                rsum += __shfl_xor(rsum, 1, 16);
                rsum += __shfl_xor(rsum, 2, 16);
                rsum += __shfl_xor(rsum, 4, 16);
                rsum += __shfl_xor(rsum, 8, 16);
                lrun[r] = lrun[r] * corr + rsum;
                mrun[r] = mnew;
#pragma unroll
                for (int dt = 0; dt < 4; ++dt) oacc[dt][r] *= corr;
            }
            // ---- P: C-layout -> A-layout via per-wave LDS patch ----
            unsigned short* myPs = Ps[wid];
#pragma unroll
            for (int tc = 0; tc < 4; ++tc)
#pragma unroll
                for (int r = 0; r < 8; ++r)
                    myPs[(8 * lh + r) * 64 + tc * 16 + l15] = f2bf(sacc[tc][r]);

            // ---- O += P V (2 K-steps x 4 d tiles = 8 WMMAs) ----
#pragma unroll
            for (int ks = 0; ks < 2; ++ks) {
                Frag aP;
                const uint4* pp = (const uint4*)(myPs + l15 * 64);
                aP.q[0] = pp[ks * 4 + lh];
                aP.q[1] = pp[ks * 4 + 2 + lh];
#pragma unroll
                for (int dt = 0; dt < 4; ++dt) {
                    Frag bV;
                    const uint4* pv = (const uint4*)(Vst + (dt * 16 + l15) * 64);
                    bV.q[0] = pv[ks * 4 + lh];
                    bV.q[1] = pv[ks * 4 + 2 + lh];
                    oacc[dt] = __builtin_amdgcn_wmma_f32_16x16x32_bf16(
                        false, aP.v, false, bV.v, (short)0, oacc[dt],
                        false, false);
                }
            }
        }
        __syncthreads();
    }

    // ---- normalize and write bf16 output (per-head slice) ----
#pragma unroll
    for (int dt = 0; dt < 4; ++dt) {
#pragma unroll
        for (int r = 0; r < 8; ++r) {
            int qr = qBase + 8 * lh + r;
            float v = oacc[dt][r] / lrun[r];
            O[(batchRow + qr) * D_MODEL + h * DK + dt * 16 + l15] = f2bf(v);
        }
    }
}

// ---------------------------------------------------------------------------
// Host-side orchestration
// ---------------------------------------------------------------------------
extern "C" void kernel_launch(void* const* d_in, const int* in_sizes, int n_in,
                              void* d_out, int out_size, void* d_ws, size_t ws_size,
                              hipStream_t stream) {
    const float* x   = (const float*)d_in[0];
    const float* w_q = (const float*)d_in[1];
    const float* w_k = (const float*)d_in[2];
    const float* w_v = (const float*)d_in[3];
    const float* w_o = (const float*)d_in[4];
    const int* tpos  = (const int*)d_in[5];
    float* out = (float*)d_out;

    // workspace carve-out (all bf16), ~64 MB total
    const size_t XW_EL = (size_t)M_TOT * D_MODEL;      // 4 M elements
    const size_t WW_EL = (size_t)D_MODEL * D_MODEL;    // 1 M elements
    char* ws = (char*)d_ws;
    size_t off = 0;
    auto carve = [&](size_t elems) {
        unsigned short* p = (unsigned short*)(ws + off);
        off += ((elems * 2) + 255) & ~(size_t)255;
        return p;
    };
    unsigned short* xw   = carve(XW_EL);
    unsigned short* wqw  = carve(WW_EL);
    unsigned short* wkw  = carve(WW_EL);
    unsigned short* wvw  = carve(WW_EL);
    unsigned short* wow  = carve(WW_EL);
    unsigned short* Qp   = carve(XW_EL);
    unsigned short* Kp   = carve(XW_EL);
    unsigned short* Vp   = carve(XW_EL);
    unsigned short* Qr   = carve(XW_EL);
    unsigned short* Kr   = carve(XW_EL);
    unsigned short* attnb = carve(XW_EL);

    // 1) f32 -> bf16 conversions
    cvt_bf16_kernel<<<(XW_EL / 4 + 255) / 256, 256, 0, stream>>>(x, xw, (int)(XW_EL / 4));
    cvt_bf16_kernel<<<(WW_EL / 4 + 255) / 256, 256, 0, stream>>>(w_q, wqw, (int)(WW_EL / 4));
    cvt_bf16_kernel<<<(WW_EL / 4 + 255) / 256, 256, 0, stream>>>(w_k, wkw, (int)(WW_EL / 4));
    cvt_bf16_kernel<<<(WW_EL / 4 + 255) / 256, 256, 0, stream>>>(w_v, wvw, (int)(WW_EL / 4));
    cvt_bf16_kernel<<<(WW_EL / 4 + 255) / 256, 256, 0, stream>>>(w_o, wow, (int)(WW_EL / 4));

    // 2) fused QKV projection: grid.z = {Q,K,V}
    {
        dim3 grid(D_MODEL / 64, M_TOT / 128, 3), block(256);
        gemm_nt_bf16<0><<<grid, block, 0, stream>>>(xw, wqw, wkw, wvw,
                                                    Qp, Kp, Vp, nullptr);
    }

    // 3) RoPE on Q and K
    {
        int total = M_TOT * NUM_HEADS * (DK / 2);  // 2,097,152
        rope_kernel<<<total / 256, 256, 0, stream>>>(Qp, Kp, Qr, Kr, tpos);
    }

    // 4) fused causal flash attention
    {
        dim3 grid(SEQ / 128, NUM_HEADS, BATCH), block(256);
        attn_kernel<<<grid, block, 0, stream>>>(Qr, Kr, Vp, attnb);
    }

    // 5) output projection -> f32 d_out
    {
        dim3 grid(D_MODEL / 64, M_TOT / 128, 1), block(256);
        gemm_nt_bf16<1><<<grid, block, 0, stream>>>(attnb, wow, wow, wow,
                                                    nullptr, nullptr, nullptr, out);
    }
}
